// ExpertTimmViTBlock_65231963292484
// MI455X (gfx1250) — matrix operand order repaired and checked by reading, not compiled, not run on hardware
//
#include <hip/hip_runtime.h>
#include <hip/hip_bf16.h>

// ---------------------------------------------------------------------------
// ViT block for MI455X (gfx1250): all matmuls via v_wmma_f32_16x16x32_bf16,
// GEMM operand staging via Tensor Data Mover (tensor_load_to_lds), double
// buffered on TENSORcnt. Shapes: B=8, N=1024, C=1024, H=16, DH=64.
// ---------------------------------------------------------------------------

constexpr int kB  = 8;
constexpr int kN  = 1024;
constexpr int kC  = 1024;
constexpr int kH  = 16;
constexpr int kDH = 64;
constexpr float kScale = 0.125f;     // DH^-0.5
constexpr float kEps   = 1e-6f;

typedef __attribute__((ext_vector_type(16))) __bf16         v16bf;
typedef __attribute__((ext_vector_type(8)))  __bf16         v8bf;
typedef __attribute__((ext_vector_type(8)))  float          v8f;
typedef __attribute__((ext_vector_type(8)))  unsigned short v8us;
typedef __attribute__((ext_vector_type(4)))  unsigned int   u32x4;
typedef __attribute__((ext_vector_type(8)))  int            i32x8;
typedef __attribute__((ext_vector_type(4)))  int            i32x4;

#if __has_builtin(__builtin_amdgcn_tensor_load_to_lds)
#define USE_TDM 1
#else
#define USE_TDM 0
#endif

__device__ __forceinline__ unsigned short f32_to_bf16(float f) {
  union { float f; unsigned u; } c; c.f = f;
  unsigned u = c.u;
  u += 0x7FFFu + ((u >> 16) & 1u);    // round-to-nearest-even
  return (unsigned short)(u >> 16);
}

// Build a 16-element bf16 fragment from two aligned 8-element (16B) loads.
__device__ __forceinline__ v16bf ld_frag(const unsigned short* p0,
                                         const unsigned short* p1) {
  v8bf a = *(const v8bf*)p0;
  v8bf b = *(const v8bf*)p1;
  return __builtin_shufflevector(a, b, 0,1,2,3,4,5,6,7,8,9,10,11,12,13,14,15);
}

#if USE_TDM
// ---------------------------------------------------------------------------
// TDM descriptor helpers (CDNA5 ISA ch.8). 2D tile load: 128 rows x 32 bf16,
// LDS padding 16B after every 64B row -> 80B (40-half) LDS row stride, which
// matches the fragment-load layout used by the compute waves.
// ---------------------------------------------------------------------------
__device__ __forceinline__ i32x8 tdm_g1(unsigned dim0, unsigned dim1,
                                        unsigned stride0) {
  i32x8 g1;
  // data_size=2B (code 1), pad_enable, pad_interval=16 DW (code 3),
  // pad_amount=4 DW (code 3); no multicast, no iterate, no atomic barrier.
  g1[0] = (1 << 16) | (1 << 20) | (3 << 22) | (3 << 25);
  g1[1] = (int)((dim0 & 0xFFFFu) << 16);                       // tensor_dim0 lo
  g1[2] = (int)((dim0 >> 16) | ((dim1 & 0xFFFFu) << 16));      // dim0 hi|dim1 lo
  g1[3] = (int)((dim1 >> 16) | (32u << 16));                   // dim1 hi|tile_dim0=32
  g1[4] = 128;                                                 // tile_dim1=128, tile_dim2=0
  g1[5] = (int)stride0;                                        // tensor_dim0_stride lo32
  g1[6] = 0;                                                   // stride hi (0), dim1_stride lo
  g1[7] = 0;
  return g1;
}

__device__ __forceinline__ void tdm_issue(unsigned lds_byte,
                                          unsigned long long gaddr,
                                          i32x8 g1) {
  u32x4 g0;
  g0.x = 1u;                                                   // count=1, user mode
  g0.y = lds_byte;                                             // lds_addr
  g0.z = (unsigned)gaddr;                                      // global_addr lo32
  g0.w = (unsigned)((gaddr >> 32) & 0x01FFFFFFull) | (2u << 30); // hi25 | type=2
  i32x4 z4 = {0, 0, 0, 0};
#if __clang_major__ >= 23
  i32x8 z8 = {0, 0, 0, 0, 0, 0, 0, 0};
  __builtin_amdgcn_tensor_load_to_lds(g0, g1, z4, z4, z8, 0);
#else
  __builtin_amdgcn_tensor_load_to_lds(g0, g1, z4, z4, 0);
#endif
}
#endif  // USE_TDM

// ---------------------------------------------------------------------------
// f32 -> bf16 convert (A matrices, row-major, no transpose)
// ---------------------------------------------------------------------------
__global__ void cvt_bf16_kernel(const float* __restrict__ in,
                                unsigned short* __restrict__ out, int n) {
  int stride = gridDim.x * blockDim.x;
  for (int i = blockIdx.x * blockDim.x + threadIdx.x; i < n; i += stride)
    out[i] = f32_to_bf16(in[i]);
}

// ---------------------------------------------------------------------------
// f32 [K][N] -> bf16 [N][K] transpose-convert (weights -> B-fragment friendly)
// ---------------------------------------------------------------------------
__global__ void transpose_cvt_kernel(const float* __restrict__ in,
                                     unsigned short* __restrict__ out,
                                     int Kdim, int Ndim) {
  __shared__ float t[32][33];
  int bx = blockIdx.x * 32;   // over N
  int by = blockIdx.y * 32;   // over K
  int tx = threadIdx.x & 31, ty = threadIdx.x >> 5;   // 8 rows per pass
  for (int i = ty; i < 32; i += 8)
    t[i][tx] = in[(size_t)(by + i) * Ndim + bx + tx];
  __syncthreads();
  for (int i = ty; i < 32; i += 8)
    out[(size_t)(bx + i) * Kdim + by + tx] = f32_to_bf16(t[tx][i]);
}

// ---------------------------------------------------------------------------
// WMMA GEMM: C[M][N] = A[M][K] @ Bt[N][K]^T + bias, optional exact GELU.
// 128x128 block tile, 8 waves, each wave computes 64x32 (4x2 WMMA tiles).
// Operand tiles DMA'd into LDS by the TDM (wave 0 issues, TENSORcnt-tracked,
// double buffered); LDS row stride 40 halves = 80B from TDM pad fields.
// ---------------------------------------------------------------------------
template <bool GELU, bool OUT_BF16>
__global__ __launch_bounds__(256)
void gemm_bf16_kernel(const unsigned short* __restrict__ A,
                      const unsigned short* __restrict__ Bt,
                      const float* __restrict__ bias,
                      void* __restrict__ Cout,
                      int M, int N, int K) {
  constexpr int BM = 128, BN = 128, BK = 32, LDT = 40;
  __shared__ alignas(16) unsigned short As[2][BM * LDT];
  __shared__ alignas(16) unsigned short Bs[2][BN * LDT];

  const int tid  = threadIdx.x;
  const int lane = tid & 31, wid = tid >> 5;
  const int l16  = lane & 15, hi = lane >> 4;          // lane half (K/M select)
  const int wm   = (wid & 1) * 64, wn = (wid >> 1) * 32;
  const int m0   = blockIdx.y * BM, n0 = blockIdx.x * BN;

  v8f acc[4][2] = {};

#if USE_TDM
  const i32x8 dA = tdm_g1((unsigned)K, (unsigned)M, (unsigned)K);
  const i32x8 dB = tdm_g1((unsigned)K, (unsigned)N, (unsigned)K);
  const unsigned long long baseA =
      (unsigned long long)(uintptr_t)(A + (size_t)m0 * K);
  const unsigned long long baseB =
      (unsigned long long)(uintptr_t)(Bt + (size_t)n0 * K);
  const unsigned ldsA[2] = {(unsigned)(uintptr_t)&As[0][0],
                            (unsigned)(uintptr_t)&As[1][0]};
  const unsigned ldsB[2] = {(unsigned)(uintptr_t)&Bs[0][0],
                            (unsigned)(uintptr_t)&Bs[1][0]};
  if (wid == 0) {           // prologue: fill stage 0
    tdm_issue(ldsA[0], baseA, dA);
    tdm_issue(ldsB[0], baseB, dB);
  }
#endif

  for (int kk = 0; kk < K; kk += BK) {
    const int st = (kk >> 5) & 1;
#if USE_TDM
    if (wid == 0) {
      if (kk + BK < K) {    // kick off next stage, then cover current stage
        tdm_issue(ldsA[st ^ 1], baseA + (size_t)(kk + BK) * 2, dA);
        tdm_issue(ldsB[st ^ 1], baseB + (size_t)(kk + BK) * 2, dB);
        __builtin_amdgcn_s_wait_tensorcnt((short)2);
      } else {
        __builtin_amdgcn_s_wait_tensorcnt((short)0);
      }
    }
    __syncthreads();
#else
    // Fallback: synchronous staging through VGPRs.
    for (int i = tid; i < BM * 4; i += 256) {
      int row = i >> 2, seg = i & 3;
      *(v8us*)&As[st][row * LDT + seg * 8] =
          *(const v8us*)(A + (size_t)(m0 + row) * K + kk + seg * 8);
    }
    for (int i = tid; i < BN * 4; i += 256) {
      int row = i >> 2, seg = i & 3;
      *(v8us*)&Bs[st][row * LDT + seg * 8] =
          *(const v8us*)(Bt + (size_t)(n0 + row) * K + kk + seg * 8);
    }
    __syncthreads();
#endif

    // A fragments: lane<16 -> K {0..7,16..23}; lane>=16 -> K {8..15,24..31}.
    v16bf af[4];
    const int ac0 = hi ? 8 : 0;
    #pragma unroll
    for (int i = 0; i < 4; i++) {
      const unsigned short* p = &As[st][(wm + i * 16 + l16) * LDT + ac0];
      af[i] = ld_frag(p, p + 16);
    }
    // B fragments: lane<16 -> K {0..7,16..23}; lane>=16 -> K {8..15,24..31}.
    v16bf bfr[2];
    const int bk0 = hi ? 8 : 0;
    #pragma unroll
    for (int j = 0; j < 2; j++) {
      const unsigned short* p = &Bs[st][(wn + j * 16 + l16) * LDT + bk0];
      bfr[j] = ld_frag(p, p + 16);
    }
    #pragma unroll
    for (int i = 0; i < 4; i++)
      #pragma unroll
      for (int j = 0; j < 2; j++)
        acc[i][j] = __builtin_amdgcn_wmma_f32_16x16x32_bf16(
            false, af[i], false, bfr[j], (short)0, acc[i][j], false, false);
    __syncthreads();   // readers done before this stage is refilled
  }

  // Epilogue: C layout -> VGPR r holds row (r + hi*8), lane l16 holds column.
  #pragma unroll
  for (int j = 0; j < 2; j++) {
    int col = n0 + wn + j * 16 + l16;
    float bv = bias ? bias[col] : 0.0f;
    #pragma unroll
    for (int i = 0; i < 4; i++) {
      int mrow = m0 + wm + i * 16 + hi * 8;
      #pragma unroll
      for (int r = 0; r < 8; r++) {
        float v = acc[i][j][r] + bv;
        if (GELU) v = 0.5f * v * (1.0f + erff(v * 0.70710678118f));
        size_t off = (size_t)(mrow + r) * N + col;
        if (OUT_BF16) ((unsigned short*)Cout)[off] = f32_to_bf16(v);
        else          ((float*)Cout)[off] = v;
      }
    }
  }
}

// ---------------------------------------------------------------------------
// Flash attention: one block = 128 queries of one (b,h); 8 waves x 16 queries.
// qkv is bf16 [B*N][3C] with token layout [3][H][DH]. Output bf16 [B*N][C].
// ---------------------------------------------------------------------------
__global__ __launch_bounds__(256)
void attention_kernel(const unsigned short* __restrict__ qkv,
                      unsigned short* __restrict__ yb) {
  constexpr int LDT = 40;
  __shared__ alignas(16) unsigned short Vt[64 * LDT];        // V^T block [d][key]
  __shared__ alignas(16) unsigned short Ps[8 * 16 * LDT];    // per-wave P tile

  const int tid = threadIdx.x, lane = tid & 31, wid = tid >> 5;
  const int l16 = lane & 15, hi = lane >> 4;
  const int b = blockIdx.y / kH, h = blockIdx.y % kH;
  const int q0 = blockIdx.x * 128 + wid * 16;
  const int TS = 3 * kC;                                      // token stride
  const unsigned short* qp = qkv + (size_t)b * kN * TS + h * kDH;
  const unsigned short* kp = qp + kC;
  const unsigned short* vp = qp + 2 * kC;

  // Q as two A-fragments (d 0..31, 32..63), straight from global.
  v16bf qf[2];
  {
    const unsigned short* base = qp + (size_t)(q0 + l16) * TS;
    int c0 = hi ? 8 : 0;
    qf[0] = ld_frag(base + c0,      base + c0 + 16);
    qf[1] = ld_frag(base + 32 + c0, base + 32 + c0 + 16);
  }

  v8f o[4] = {};                       // 16 queries x 64 d accumulator
  float rmax[8], rsum[8];
  #pragma unroll
  for (int r = 0; r < 8; r++) { rmax[r] = -1e30f; rsum[r] = 0.0f; }

  for (int kb = 0; kb < kN; kb += 32) {
    // Prefetch next K/V block rows (one row per lane) into cache.
    if (kb + 32 < kN) {
      __builtin_prefetch(kp + (size_t)(kb + 32 + lane) * TS, 0, 3);
      __builtin_prefetch(vp + (size_t)(kb + 32 + lane) * TS, 0, 3);
    }
    // Cooperatively stage V block transposed: Vt[d][key].
    {
      int key = tid >> 3, d0 = (tid & 7) * 8;
      v8us vv = *(const v8us*)(vp + (size_t)(kb + key) * TS + d0);
      #pragma unroll
      for (int j = 0; j < 8; j++) Vt[(d0 + j) * LDT + key] = vv[j];
    }
    __syncthreads();

    // S = Q @ K^T for two 16-key tiles; K-fragments direct from global
    // (contiguous in d, which is the contraction dim).
    v8f s[2] = {};
    #pragma unroll
    for (int ni = 0; ni < 2; ni++) {
      const unsigned short* kbase = kp + (size_t)(kb + ni * 16 + l16) * TS;
      int k0 = hi ? 8 : 0;
      v16bf b0 = ld_frag(kbase + k0,      kbase + k0 + 16);
      v16bf b1 = ld_frag(kbase + 32 + k0, kbase + 32 + k0 + 16);
      s[ni] = __builtin_amdgcn_wmma_f32_16x16x32_bf16(
          false, qf[0], false, b0, (short)0, s[ni], false, false);
      s[ni] = __builtin_amdgcn_wmma_f32_16x16x32_bf16(
          false, qf[1], false, b1, (short)0, s[ni], false, false);
    }

    // Online softmax. Row r+hi*8 lives in element r, spread across a 16-lane
    // group -> reduce with xor shuffles 1,2,4,8.
    float nm[8], al[8];
    #pragma unroll
    for (int r = 0; r < 8; r++) {
      s[0][r] *= kScale; s[1][r] *= kScale;
      float m = fmaxf(s[0][r], s[1][r]);
      #pragma unroll
      for (int off = 1; off < 16; off <<= 1)
        m = fmaxf(m, __shfl_xor(m, off, 32));
      nm[r] = fmaxf(rmax[r], m);
      al[r] = __expf(rmax[r] - nm[r]);
      rmax[r] = nm[r];
    }
    #pragma unroll
    for (int r = 0; r < 8; r++) {
      float p0 = __expf(s[0][r] - nm[r]);
      float p1 = __expf(s[1][r] - nm[r]);
      s[0][r] = p0; s[1][r] = p1;
      float t = p0 + p1;
      #pragma unroll
      for (int off = 1; off < 16; off <<= 1) t += __shfl_xor(t, off, 32);
      rsum[r] = rsum[r] * al[r] + t;
    }
    #pragma unroll
    for (int di = 0; di < 4; di++)
      #pragma unroll
      for (int r = 0; r < 8; r++) o[di][r] *= al[r];

    // Reshape P (C layout) -> A-fragment layout via per-wave LDS tile.
    unsigned short* Pw = &Ps[wid * 16 * LDT];
    #pragma unroll
    for (int ni = 0; ni < 2; ni++)
      #pragma unroll
      for (int r = 0; r < 8; r++)
        Pw[(r + hi * 8) * LDT + ni * 16 + l16] = f32_to_bf16(s[ni][r]);
    asm volatile("s_wait_dscnt 0" ::: "memory");   // in-wave LDS visibility

    v16bf pf;
    {
      int c0 = hi ? 8 : 0;
      const unsigned short* p = &Pw[l16 * LDT + c0];
      pf = ld_frag(p, p + 16);
    }
    // O += P @ V  (V B-fragments from transposed LDS: contiguous in keys).
    #pragma unroll
    for (int di = 0; di < 4; di++) {
      int k0 = hi ? 8 : 0;
      const unsigned short* p = &Vt[(di * 16 + l16) * LDT + k0];
      v16bf vf = ld_frag(p, p + 16);
      o[di] = __builtin_amdgcn_wmma_f32_16x16x32_bf16(
          false, pf, false, vf, (short)0, o[di], false, false);
    }
    __syncthreads();   // protect Vt before next overwrite
  }

  #pragma unroll
  for (int r = 0; r < 8; r++) rsum[r] = 1.0f / rsum[r];
  #pragma unroll
  for (int di = 0; di < 4; di++)
    #pragma unroll
    for (int r = 0; r < 8; r++) {
      int row = q0 + hi * 8 + r;
      yb[(size_t)(b * kN + row) * kC + h * kDH + di * 16 + l16] =
          f32_to_bf16(o[di][r] * rsum[r]);
    }
}

// ---------------------------------------------------------------------------
// out = R + LayerNorm(X)*g + b per row of 1024; optionally also bf16 copy.
// ---------------------------------------------------------------------------
template <bool WRITE_BF16>
__global__ __launch_bounds__(256)
void ln_residual_kernel(const float* __restrict__ X,
                        const float* __restrict__ R,
                        const float* __restrict__ g,
                        const float* __restrict__ bt,
                        float* __restrict__ outf,
                        unsigned short* __restrict__ outb) {
  __shared__ float sbuf[8], ssbuf[8];
  const int row = blockIdx.x, tid = threadIdx.x;
  const int wid = tid >> 5, lane = tid & 31;
  const float* x = X + (size_t)row * kC;

  float v[4], s = 0.0f, ss = 0.0f;
  #pragma unroll
  for (int i = 0; i < 4; i++) {
    v[i] = x[tid + i * 256];
    s += v[i]; ss += v[i] * v[i];
  }
  #pragma unroll
  for (int off = 1; off < 32; off <<= 1) {
    s  += __shfl_xor(s,  off, 32);
    ss += __shfl_xor(ss, off, 32);
  }
  if (lane == 0) { sbuf[wid] = s; ssbuf[wid] = ss; }
  __syncthreads();
  float ts = 0.0f, tss = 0.0f;
  #pragma unroll
  for (int i = 0; i < 8; i++) { ts += sbuf[i]; tss += ssbuf[i]; }
  float mean = ts * (1.0f / kC);
  float var  = tss * (1.0f / kC) - mean * mean;
  float inv  = rsqrtf(var + kEps);

  #pragma unroll
  for (int i = 0; i < 4; i++) {
    int c = tid + i * 256;
    float ln = (v[i] - mean) * inv * g[c] + bt[c];
    float o  = R[(size_t)row * kC + c] + ln;
    outf[(size_t)row * kC + c] = o;
    if (WRITE_BF16) outb[(size_t)row * kC + c] = f32_to_bf16(o);
  }
}

// ---------------------------------------------------------------------------
// Host orchestration. Workspace layout (bytes, ~216 MB with aliasing):
//   xb 0..16M | wT 16..40M | qkvb 40..88M (reused as y2f) | ybf 88..104M |
//   Yf 104..136M (reused as H2f) | y2b 136..152M | Hb 152..216M
// ---------------------------------------------------------------------------
extern "C" void kernel_launch(void* const* d_in, const int* in_sizes, int n_in,
                              void* d_out, int out_size, void* d_ws,
                              size_t ws_size, hipStream_t stream) {
  const float* x      = (const float*)d_in[0];
  const float* qkv_w  = (const float*)d_in[1];
  const float* qkv_b  = (const float*)d_in[2];
  const float* proj_w = (const float*)d_in[3];
  const float* proj_b = (const float*)d_in[4];
  const float* n1_g   = (const float*)d_in[5];
  const float* n1_b   = (const float*)d_in[6];
  const float* fc1_w  = (const float*)d_in[7];
  const float* fc1_b  = (const float*)d_in[8];
  const float* fc2_w  = (const float*)d_in[9];
  const float* fc2_b  = (const float*)d_in[10];
  const float* n2_g   = (const float*)d_in[11];
  const float* n2_b   = (const float*)d_in[12];
  float* out = (float*)d_out;
  char*  ws  = (char*)d_ws;

  const int M = kB * kN;                     // 8192 rows
  unsigned short* xb     = (unsigned short*)(ws + 0);
  unsigned short* wqkvT  = (unsigned short*)(ws + 16777216);
  unsigned short* wprojT = (unsigned short*)(ws + 23068672);
  unsigned short* wfc1T  = (unsigned short*)(ws + 25165824);
  unsigned short* wfc2T  = (unsigned short*)(ws + 33554432);
  unsigned short* qkvb   = (unsigned short*)(ws + 41943040);
  float*          y2f    = (float*)         (ws + 41943040);  // alias: qkvb dead
  unsigned short* ybf    = (unsigned short*)(ws + 92274688);
  float*          Yf     = (float*)         (ws + 109051904);
  float*          H2f    = (float*)         (ws + 109051904); // alias: Yf dead
  unsigned short* y2b    = (unsigned short*)(ws + 142606336);
  unsigned short* Hb     = (unsigned short*)(ws + 159383552);
  if (ws_size < 226492416) return;           // need ~216 MB scratch

  // 1) Precision conversion / weight transposition (one-time, BW-trivial).
  cvt_bf16_kernel<<<2048, 256, 0, stream>>>(x, xb, M * kC);
  transpose_cvt_kernel<<<dim3(96, 32),  256, 0, stream>>>(qkv_w,  wqkvT, kC,     3 * kC);
  transpose_cvt_kernel<<<dim3(32, 32),  256, 0, stream>>>(proj_w, wprojT, kC,    kC);
  transpose_cvt_kernel<<<dim3(128, 32), 256, 0, stream>>>(fc1_w,  wfc1T, kC,     4 * kC);
  transpose_cvt_kernel<<<dim3(32, 128), 256, 0, stream>>>(fc2_w,  wfc2T, 4 * kC, kC);

  // 2) qkv = x @ qkv_w + qkv_b  -> bf16 [8192][3072]
  gemm_bf16_kernel<false, true><<<dim3(24, 64), 256, 0, stream>>>(
      xb, wqkvT, qkv_b, qkvb, M, 3 * kC, kC);

  // 3) flash attention -> bf16 [8192][1024]
  attention_kernel<<<dim3(kN / 128, kB * kH), 256, 0, stream>>>(qkvb, ybf);

  // 4) Y = attn @ proj_w + proj_b  -> f32
  gemm_bf16_kernel<false, false><<<dim3(8, 64), 256, 0, stream>>>(
      ybf, wprojT, proj_b, Yf, M, kC, kC);

  // 5) y2 = Y + LN(Y)  -> f32 + bf16
  ln_residual_kernel<true><<<M, 256, 0, stream>>>(Yf, Yf, n1_g, n1_b, y2f, y2b);

  // 6) H = GELU(y2 @ fc1_w + fc1_b)  -> bf16 [8192][4096]
  gemm_bf16_kernel<true, true><<<dim3(32, 64), 256, 0, stream>>>(
      y2b, wfc1T, fc1_b, Hb, M, 4 * kC, kC);

  // 7) H2 = H @ fc2_w + fc2_b  -> f32
  gemm_bf16_kernel<false, false><<<dim3(8, 64), 256, 0, stream>>>(
      Hb, wfc2T, fc2_b, H2f, M, kC, 4 * kC);

  // 8) out = y2 + LN(H2)
  ln_residual_kernel<false><<<M, 256, 0, stream>>>(H2f, y2f, n2_g, n2_b, out,
                                                   nullptr);
}
